// StochasticAttention_27230092656804
// MI455X (gfx1250) — matrix operand order repaired and checked by reading, not compiled, not run on hardware
//
#include <hip/hip_runtime.h>

// ---------------------------------------------------------------------------
// StochasticAttention, MI455X (gfx1250).
//
// Algebraic collapse: out[b,q,v] = sum_s softmax(scores')[b,q,s] * sum_d V[b,v,d]
//                               = 1 * rowsum_d(V)[b,v]      (exact identity)
// where V = x @ Wv^T. The sampling/scatter/softmax path cannot affect the
// output. Essential work: read x (64 MB) + write out (64 MB) -> ~5.5 us at
// 23.3 TB/s. The V GEMM (2.1 GFLOP) rides along on the WMMA pipe for free.
//
// Stage 1: vsum[16384] = rowsum_d( X[16384x1024] @ Wv^T[1024x64] )
//          via v_wmma_f32_16x16x32_bf16 with 3xBF16 splitting (hi/lo) for
//          ~fp32 accuracy. Wv packed as bf16 B-fragments in LDS.
// Stage 2: out[b,q,:] = vsum[b,:] broadcast, coalesced float4 stores.
// ---------------------------------------------------------------------------

#define F_DIM 1024
#define D_DIM 64
#define B_DIM 16
#define M_TOTAL (B_DIM * F_DIM)        // 16384 GEMM rows
#define ROWS_PER_BLOCK 64              // 4 waves * 16 rows
#define GEMM_BLOCKS (M_TOTAL / ROWS_PER_BLOCK)

typedef __attribute__((ext_vector_type(16))) __bf16          v16bf;
typedef __attribute__((ext_vector_type(16))) unsigned short  v16us;
typedef __attribute__((ext_vector_type(8)))  float           v8f;
typedef __attribute__((ext_vector_type(8)))  unsigned int    v8u;

static_assert(sizeof(v16bf) == 32 && sizeof(v16us) == 32 && sizeof(v8u) == 32, "frag size");

__device__ __forceinline__ unsigned short f2bf(float f) {
  union { float f; unsigned u; } c; c.f = f;
  unsigned u = c.u + 0x7FFFu + ((c.u >> 16) & 1u);   // round-to-nearest-even
  return (unsigned short)(u >> 16);
}
__device__ __forceinline__ float bf2f(unsigned short h) {
  union { unsigned u; float f; } c; c.u = (unsigned)h << 16;
  return c.f;
}
// split f32 into bf16 hi + bf16 lo (residual), so hi*hi+hi*lo+lo*hi ~ fp32 product
__device__ __forceinline__ void split_bf(float f, unsigned short& hi, unsigned short& lo) {
  hi = f2bf(f);
  lo = f2bf(f - bf2f(hi));
}

// ---------------------------------------------------------------------------
// Stage 1: vsum = rowsum_d( X @ Wv^T ), bf16x3 WMMA.
// Block = 128 threads = 4 waves; each wave owns a 16-row M tile, all 4 N tiles
// (D=64 = 4x16). K=1024 processed in 4 phases of 256; per phase the block
// packs Wv[:, phase] into LDS as hi/lo bf16 B-fragments (64 KB total LDS).
// ---------------------------------------------------------------------------
__global__ void __launch_bounds__(128)
vrowsum_wmma(const float* __restrict__ x, const float* __restrict__ wv,
             float* __restrict__ vsum) {
  // per phase: 4 ntiles * 8 ksteps * 32 lanes entries, 8 packed uints each
  __shared__ unsigned int ldsBhi[8192];   // 32 KB
  __shared__ unsigned int ldsBlo[8192];   // 32 KB

  const int tid   = threadIdx.x;
  const int wave  = tid >> 5;
  const int lane  = tid & 31;
  const int laneM = lane & 15;     // M row within tile (A layout), N column (C layout)
  const int laneG = lane >> 4;     // K-group half (A layout), M-group half (C layout)
  const int mBase = blockIdx.x * ROWS_PER_BLOCK + wave * 16;

  v8f acc[4];
#pragma unroll
  for (int t = 0; t < 4; ++t)
#pragma unroll
    for (int r = 0; r < 8; ++r) acc[t][r] = 0.0f;

  const float* arow = x + (size_t)(mBase + laneM) * F_DIM;

#pragma unroll 1
  for (int phase = 0; phase < 4; ++phase) {
    const int kQ = phase * 256;

    // ---- cooperative pack of Wv[:, kQ : kQ+256] into hi/lo bf16 fragments ----
    for (int e = tid; e < 1024; e += 128) {
      const int pl = e & 31;            // consuming lane
      const int ks = (e >> 5) & 7;      // kstep within phase
      const int nt = e >> 8;            // ntile
      // B-fragment convention: lane holds N = pl&15; uint j packs K = (j>>2)*16 + (pl>>4)*8 + (j&3)*2 (+1)
      const int kcol = kQ + ks * 32 + ((pl >> 4) << 3);
      const int nRow = nt * 16 + (pl & 15);
      unsigned int* dh = &ldsBhi[e * 8];
      unsigned int* dl = &ldsBlo[e * 8];
#pragma unroll
      for (int j = 0; j < 8; ++j) {
        unsigned short h0, l0, h1, l1;
        const int k = kcol + ((j >> 2) << 4) + ((j & 3) << 1);
        split_bf(wv[(size_t)nRow * F_DIM + k],     h0, l0);
        split_bf(wv[(size_t)nRow * F_DIM + k + 1], h1, l1);
        dh[j] = (unsigned)h0 | ((unsigned)h1 << 16);
        dl[j] = (unsigned)l0 | ((unsigned)l1 << 16);
      }
    }
    __syncthreads();

    // ---- K loop over this phase ----
#pragma unroll 1
    for (int ks = 0; ks < 8; ++ks) {
      const int k0 = kQ + ks * 32;

      // A fragment (16x32 bf16): VGPR j holds K = (j>>2)*16 + laneG*8 + (j&3)*2 (+1)
      v16us auh, aul;
#pragma unroll
      for (int jj = 0; jj < 8; ++jj) {
        const int k = k0 + ((jj >> 2) << 4) + (laneG << 3) + ((jj & 3) << 1);
        const float2 f2 = *(const float2*)(arow + k);
        unsigned short h, l;
        split_bf(f2.x, h, l); auh[2 * jj]     = h; aul[2 * jj]     = l;
        split_bf(f2.y, h, l); auh[2 * jj + 1] = h; aul[2 * jj + 1] = l;
      }
      const v16bf a_hi = __builtin_bit_cast(v16bf, auh);
      const v16bf a_lo = __builtin_bit_cast(v16bf, aul);

#pragma unroll
      for (int nt = 0; nt < 4; ++nt) {
        const int idx = (((nt << 3) | ks) * 32 + lane) * 8;
        const v16bf b_hi = __builtin_bit_cast(v16bf, *(const v8u*)&ldsBhi[idx]);
        const v16bf b_lo = __builtin_bit_cast(v16bf, *(const v8u*)&ldsBlo[idx]);
        // 3xBF16 product: hi*hi + hi*lo + lo*hi  (~fp32 fidelity)
        acc[nt] = __builtin_amdgcn_wmma_f32_16x16x32_bf16(
            false, a_hi, false, b_hi, (short)0, acc[nt], false, false);
        acc[nt] = __builtin_amdgcn_wmma_f32_16x16x32_bf16(
            false, a_hi, false, b_lo, (short)0, acc[nt], false, false);
        acc[nt] = __builtin_amdgcn_wmma_f32_16x16x32_bf16(
            false, a_lo, false, b_hi, (short)0, acc[nt], false, false);
      }
    }
    __syncthreads();
  }

  // ---- reduce C over N (= D). C layout: VGPR r -> M = r + 8*laneG, N = laneM.
  v8f s;
#pragma unroll
  for (int r = 0; r < 8; ++r)
    s[r] = acc[0][r] + acc[1][r] + acc[2][r] + acc[3][r];
  // butterfly over the 16 N-lanes within each half-wave
#pragma unroll
  for (int m = 1; m <= 8; m <<= 1) {
#pragma unroll
    for (int r = 0; r < 8; ++r) s[r] += __shfl_xor(s[r], m, 32);
  }
  if (laneM == 0) {
#pragma unroll
    for (int r = 0; r < 8; ++r)
      vsum[mBase + (laneG << 3) + r] = s[r];
  }
}

// ---------------------------------------------------------------------------
// Stage 2: out[b, q, :] = vsum[b, :] for all q. Pure streaming-store stage.
// Block: 256 threads, one float4 (16 B) per thread, reused for 16 q-rows.
// grid = (F/16, B) = (64, 16). vsum (64 KB) stays L2-resident.
// ---------------------------------------------------------------------------
__global__ void __launch_bounds__(256)
bcast_rows(const float* __restrict__ vsum, float* __restrict__ out) {
  const int b   = blockIdx.y;
  const int q0  = blockIdx.x * 16;
  const int tid = threadIdx.x;
  const float4 val = ((const float4*)(vsum + (size_t)b * F_DIM))[tid];
  float4* dst = (float4*)(out + ((size_t)b * F_DIM + q0) * F_DIM) + tid;
#pragma unroll
  for (int q = 0; q < 16; ++q)
    dst[q * (F_DIM / 4)] = val;
}

// ---------------------------------------------------------------------------
// Inputs (setup_inputs order): 0:x 1:scores 2:w_query 3:w_key 4:w_value 5:idx
// scores / w_query / w_key / idx are provably dead (softmax rowsum == 1).
// ---------------------------------------------------------------------------
extern "C" void kernel_launch(void* const* d_in, const int* in_sizes, int n_in,
                              void* d_out, int out_size, void* d_ws, size_t ws_size,
                              hipStream_t stream) {
  const float* x  = (const float*)d_in[0];
  const float* wv = (const float*)d_in[4];
  float* out  = (float*)d_out;
  float* vsum = (float*)d_ws;                 // 16384 floats = 64 KB scratch

  vrowsum_wmma<<<dim3(GEMM_BLOCKS), dim3(128), 0, stream>>>(x, wv, vsum);
  bcast_rows<<<dim3(F_DIM / 16, B_DIM), dim3(256), 0, stream>>>(vsum, out);
}
